// CachedMultiHeadAttention_27719718928942
// MI455X (gfx1250) — matrix operand order repaired and checked by reading, not compiled, not run on hardware
//
#include <hip/hip_runtime.h>
#include <hip/hip_bf16.h>
#include <stdint.h>

#define SEQ    2048
#define EMB    2048
#define NHEADS 16
#define DH     128
#define BATCH  2

typedef __attribute__((ext_vector_type(16))) __bf16         v16bf;
typedef __attribute__((ext_vector_type(8)))  float          v8f;
typedef __attribute__((ext_vector_type(8)))  unsigned short u16x8;
typedef __attribute__((ext_vector_type(16))) unsigned short u16x16;

union FragAB {
  v16bf  bf;
  u16x16 w;
  u16x8  h[2];
};

__device__ __forceinline__ unsigned short f2bf(float f) {
  unsigned u = __float_as_uint(f);
  u += 0x7FFFu + ((u >> 16) & 1u);   // round-to-nearest-even
  return (unsigned short)(u >> 16);
}

// ---------------------------------------------------------------------------
// f32 -> bf16 conversion
// ---------------------------------------------------------------------------
__global__ void f32_to_bf16_kernel(const float* __restrict__ in,
                                   unsigned short* __restrict__ out, int n) {
  for (int i = blockIdx.x * blockDim.x + threadIdx.x; i < n;
       i += gridDim.x * blockDim.x)
    out[i] = f2bf(in[i]);
}

// ---------------------------------------------------------------------------
// GEMM: Y[m,n] = sum_k A[m,k] * W[n,k] + bias[n]
// A: [M][K] bf16 (row-major), W: [N][K] bf16 (row n contiguous in k)
// 32x32 tile per wave: 2 A-frags + 2 B-frags -> 4 WMMAs per 32-wide k-step
// (32 B/lane of fragment traffic per WMMA vs 64 B/lane for a 16x16 tile).
// Block = 8 waves arranged 4(M) x 2(N) -> 128x64 block tile.
// MODE 0: f32 row-major output (+bias)            -> final projection
// MODE 1: bf16 head layout [B,H,SEQ,DH] (+bias)   -> Q, K
// MODE 2: bf16 transposed head layout [B,H,DH,SEQ] (+bias) -> V
// ---------------------------------------------------------------------------
template <int MODE>
__global__ __launch_bounds__(256)
void gemm_bf16_kernel(const unsigned short* __restrict__ A,
                      const unsigned short* __restrict__ W,
                      const float* __restrict__ bias,
                      void* __restrict__ out, int M, int N, int K) {
  const int lane = threadIdx.x & 31;
  const int wave = threadIdx.x >> 5;
  const int hl   = lane >> 4;   // lane half (0/1)
  const int l16  = lane & 15;

  const int mTile = blockIdx.y * 128 + (wave & 3) * 32;
  const int nTile = blockIdx.x * 64  + (wave >> 2) * 32;

  const unsigned short* arow0 = A + (size_t)(mTile + l16)      * K;
  const unsigned short* arow1 = A + (size_t)(mTile + 16 + l16) * K;
  const unsigned short* brow0 = W + (size_t)(nTile + l16)      * K;
  const unsigned short* brow1 = W + (size_t)(nTile + 16 + l16) * K;

  v8f acc00 = {}, acc01 = {}, acc10 = {}, acc11 = {};
  for (int k0 = 0; k0 < K; k0 += 32) {
    FragAB a0, a1, b0, b1;
    // A 16x32 fragments: lane half selects 16 contiguous K [hl*16, hl*16+16)
    a0.h[0] = *(const u16x8*)(arow0 + k0 + hl * 16);
    a0.h[1] = *(const u16x8*)(arow0 + k0 + hl * 16 + 8);
    a1.h[0] = *(const u16x8*)(arow1 + k0 + hl * 16);
    a1.h[1] = *(const u16x8*)(arow1 + k0 + hl * 16 + 8);
    // B 32x16 fragments: lane holds column n=l16, 16 contiguous K per half
    b0.w = *(const u16x16*)(brow0 + k0 + hl * 16);
    b1.w = *(const u16x16*)(brow1 + k0 + hl * 16);
    acc00 = __builtin_amdgcn_wmma_f32_16x16x32_bf16(false, a0.bf, false, b0.bf,
                                                    (short)0, acc00, false, false);
    acc01 = __builtin_amdgcn_wmma_f32_16x16x32_bf16(false, a0.bf, false, b1.bf,
                                                    (short)0, acc01, false, false);
    acc10 = __builtin_amdgcn_wmma_f32_16x16x32_bf16(false, a1.bf, false, b0.bf,
                                                    (short)0, acc10, false, false);
    acc11 = __builtin_amdgcn_wmma_f32_16x16x32_bf16(false, a1.bf, false, b1.bf,
                                                    (short)0, acc11, false, false);
  }

  const float bn0 = bias[nTile + l16];
  const float bn1 = bias[nTile + 16 + l16];

#pragma unroll
  for (int mi = 0; mi < 2; ++mi) {
#pragma unroll
    for (int ni = 0; ni < 2; ++ni) {
      const v8f c = (mi == 0) ? (ni == 0 ? acc00 : acc01)
                              : (ni == 0 ? acc10 : acc11);
      const float bn = ni == 0 ? bn0 : bn1;
      const int n = nTile + ni * 16 + l16;
#pragma unroll
      for (int r = 0; r < 8; ++r) {
        const int m = mTile + mi * 16 + hl * 8 + r;
        const float y = c[r] + bn;
        if (MODE == 0) {
          ((float*)out)[(size_t)m * N + n] = y;
        } else {
          const int bb = m >> 11, q = m & (SEQ - 1);
          const int hh = n >> 7,  d = n & (DH - 1);
          unsigned short* o = (unsigned short*)out;
          if (MODE == 1)
            o[(((size_t)(bb * NHEADS + hh)) * SEQ + q) * DH + d] = f2bf(y);
          else
            o[(((size_t)(bb * NHEADS + hh)) * DH + d) * SEQ + q] = f2bf(y);
        }
      }
    }
  }
}

// ---------------------------------------------------------------------------
// Flash attention (causal), online softmax.
// One wave handles a 16-row query tile; key tiles of 32.
// Q,K: [B*H][SEQ][DH] bf16;  V: [B*H][DH][SEQ] bf16 (transposed)
// ctx: [B][SEQ][EMB] bf16
// ---------------------------------------------------------------------------
__global__ __launch_bounds__(256)
void flash_attn_kernel(const unsigned short* __restrict__ Q,
                       const unsigned short* __restrict__ Kd,
                       const unsigned short* __restrict__ Vt,
                       unsigned short* __restrict__ ctx) {
  __shared__ unsigned short pstage[8][16][32];   // per-wave P staging (1 KB each)

  const int lane = threadIdx.x & 31;
  const int wave = threadIdx.x >> 5;
  const int hl   = lane >> 4;
  const int l16  = lane & 15;

  const int bh    = blockIdx.x;                  // 0..B*H-1
  const int qTile = blockIdx.y * 8 + wave;
  const int qBase = qTile * 16;

  const unsigned short* Qp = Q  + (size_t)bh * SEQ * DH;
  const unsigned short* Kp = Kd + (size_t)bh * SEQ * DH;
  const unsigned short* Vp = Vt + (size_t)bh * DH * SEQ;

  // Load Q tile as 4 A-fragments (16x32 each, covering Dh=128)
  FragAB qa[4];
  const unsigned short* qrow = Qp + (size_t)(qBase + l16) * DH;
#pragma unroll
  for (int dc = 0; dc < 4; ++dc) {
    qa[dc].h[0] = *(const u16x8*)(qrow + dc * 32 + hl * 16);
    qa[dc].h[1] = *(const u16x8*)(qrow + dc * 32 + hl * 16 + 8);
  }

  const float scale = 0.08838834764831845f;      // 1/sqrt(128)

  float mrow[8], lrow[8], alpha[8], p0[8], p1[8];
  v8f acc[8];
#pragma unroll
  for (int r = 0; r < 8; ++r) { mrow[r] = -1e30f; lrow[r] = 0.f; }
#pragma unroll
  for (int t = 0; t < 8; ++t) { v8f z = {}; acc[t] = z; }

  for (int kBase = 0; kBase <= qBase + 15; kBase += 32) {
    // ---- S = Q * K^T for 32 keys (two 16x16 C tiles) ----
    v8f s0 = {}, s1 = {};
#pragma unroll
    for (int dc = 0; dc < 4; ++dc) {
      FragAB kb0, kb1;
      kb0.w = *(const u16x16*)(Kp + (size_t)(kBase + l16)      * DH + dc * 32 + hl * 16);
      kb1.w = *(const u16x16*)(Kp + (size_t)(kBase + 16 + l16) * DH + dc * 32 + hl * 16);
      s0 = __builtin_amdgcn_wmma_f32_16x16x32_bf16(false, qa[dc].bf, false, kb0.bf,
                                                   (short)0, s0, false, false);
      s1 = __builtin_amdgcn_wmma_f32_16x16x32_bf16(false, qa[dc].bf, false, kb1.bf,
                                                   (short)0, s1, false, false);
    }

    // ---- scale, causal mask, online softmax ----
#pragma unroll
    for (int r = 0; r < 8; ++r) {
      const int qg  = qBase + hl * 8 + r;
      const int kg0 = kBase + l16;
      const int kg1 = kBase + 16 + l16;
      const float v0 = (kg0 <= qg) ? s0[r] * scale : -1e30f;
      const float v1 = (kg1 <= qg) ? s1[r] * scale : -1e30f;
      float rm = fmaxf(v0, v1);                  // row max across 16 lanes of half
      rm = fmaxf(rm, __shfl_xor(rm, 1));
      rm = fmaxf(rm, __shfl_xor(rm, 2));
      rm = fmaxf(rm, __shfl_xor(rm, 4));
      rm = fmaxf(rm, __shfl_xor(rm, 8));
      const float mnew = fmaxf(mrow[r], rm);
      alpha[r] = __expf(mrow[r] - mnew);
      mrow[r]  = mnew;
      p0[r] = __expf(v0 - mnew);
      p1[r] = __expf(v1 - mnew);
      float rs = p0[r] + p1[r];
      rs += __shfl_xor(rs, 1);
      rs += __shfl_xor(rs, 2);
      rs += __shfl_xor(rs, 4);
      rs += __shfl_xor(rs, 8);
      lrow[r] = lrow[r] * alpha[r] + rs;
    }
#pragma unroll
    for (int t = 0; t < 8; ++t)
#pragma unroll
      for (int r = 0; r < 8; ++r) acc[t][r] *= alpha[r];

    // ---- C-layout -> A-layout transpose of P via LDS ----
#pragma unroll
    for (int r = 0; r < 8; ++r) {
      pstage[wave][hl * 8 + r][l16]      = f2bf(p0[r]);
      pstage[wave][hl * 8 + r][16 + l16] = f2bf(p1[r]);
    }
    asm volatile("s_wait_dscnt 0" ::: "memory");  // wave-local LDS RAW fence
    FragAB pa;
    {
      const unsigned short* prow = &pstage[wave][l16][0];
      pa.h[0] = *(const u16x8*)(prow + hl * 16);
      pa.h[1] = *(const u16x8*)(prow + hl * 16 + 8);
    }

    // ---- O += P * V  (V transposed: contiguous B fragments) ----
#pragma unroll
    for (int t = 0; t < 8; ++t) {
      FragAB vb;
      vb.w = *(const u16x16*)(Vp + (size_t)(t * 16 + l16) * SEQ + kBase + hl * 16);
      acc[t] = __builtin_amdgcn_wmma_f32_16x16x32_bf16(false, pa.bf, false, vb.bf,
                                                       (short)0, acc[t], false, false);
    }
  }

  // ---- normalize and store context [B][SEQ][EMB] ----
  const int bb = bh >> 4;
  const int hh = bh & 15;
#pragma unroll
  for (int r = 0; r < 8; ++r) {
    const int q = qBase + hl * 8 + r;
    const float inv = 1.0f / lrow[r];
#pragma unroll
    for (int t = 0; t < 8; ++t) {
      const int d = t * 16 + l16;
      ctx[((size_t)(bb * SEQ + q)) * EMB + hh * DH + d] = f2bf(acc[t][r] * inv);
    }
  }
}

// ---------------------------------------------------------------------------
extern "C" void kernel_launch(void* const* d_in, const int* in_sizes, int n_in,
                              void* d_out, int out_size, void* d_ws, size_t ws_size,
                              hipStream_t stream) {
  (void)in_sizes; (void)n_in; (void)out_size; (void)ws_size;
  const float* x  = (const float*)d_in[0];
  const float* wq = (const float*)d_in[1];
  const float* bq = (const float*)d_in[2];
  const float* wk = (const float*)d_in[3];
  const float* bk = (const float*)d_in[4];
  const float* wv = (const float*)d_in[5];
  const float* bv = (const float*)d_in[6];
  const float* wo = (const float*)d_in[7];
  const float* bo = (const float*)d_in[8];
  float* out = (float*)d_out;

  // workspace layout (bytes)
  char* ws = (char*)d_ws;
  unsigned short* xb  = (unsigned short*)(ws);                 // 16 MiB
  unsigned short* wqb = (unsigned short*)(ws + 16777216ull);   // 8 MiB each
  unsigned short* wkb = (unsigned short*)(ws + 25165824ull);
  unsigned short* wvb = (unsigned short*)(ws + 33554432ull);
  unsigned short* wob = (unsigned short*)(ws + 41943040ull);
  unsigned short* Qh  = (unsigned short*)(ws + 50331648ull);   // 16 MiB each
  unsigned short* Kh  = (unsigned short*)(ws + 67108864ull);
  unsigned short* Vt  = (unsigned short*)(ws + 83886080ull);
  unsigned short* ctx = (unsigned short*)(ws + 100663296ull);  // end @ 112 MiB

  const int NX = BATCH * SEQ * EMB;  // 8388608
  const int NW = EMB * EMB;          // 4194304
  f32_to_bf16_kernel<<<2048, 256, 0, stream>>>(x,  xb,  NX);
  f32_to_bf16_kernel<<<1024, 256, 0, stream>>>(wq, wqb, NW);
  f32_to_bf16_kernel<<<1024, 256, 0, stream>>>(wk, wkb, NW);
  f32_to_bf16_kernel<<<1024, 256, 0, stream>>>(wv, wvb, NW);
  f32_to_bf16_kernel<<<1024, 256, 0, stream>>>(wo, wob, NW);

  const int M = BATCH * SEQ;         // 4096
  dim3 ggrid(EMB / 64, M / 128);     // (32, 32): 128x64 block tile, 32x32/wave
  gemm_bf16_kernel<1><<<ggrid, 256, 0, stream>>>(xb, wqb, bq, Qh, M, EMB, EMB);
  gemm_bf16_kernel<1><<<ggrid, 256, 0, stream>>>(xb, wkb, bk, Kh, M, EMB, EMB);
  gemm_bf16_kernel<2><<<ggrid, 256, 0, stream>>>(xb, wvb, bv, Vt, M, EMB, EMB);

  dim3 agrid(BATCH * NHEADS, SEQ / 16 / 8);   // (32, 16)
  flash_attn_kernel<<<agrid, 256, 0, stream>>>(Qh, Kh, Vt, ctx);

  gemm_bf16_kernel<0><<<ggrid, 256, 0, stream>>>(ctx, wob, bo, out, M, EMB, EMB);
}
